// GridBasedStructModule_65103114273342
// MI455X (gfx1250) — matrix operand order repaired and checked by reading, not compiled, not run on hardware
//
#include <hip/hip_runtime.h>
#include <hip/hip_bf16.h>
#include <math.h>

// ---------------------------------------------------------------------------
// GridBasedStructModule forward for MI455X (gfx1250, wave32, WMMA).
//
// Problem sizes (fixed by the reference):
//   D = 128, B = 32, V = 8192, A = 64, N = B*V = 262144
// Outputs (concatenated in d_out, float32):
//   predicted_coords  [B, A, 3]   = 6144 floats
//   attention_weights [B, A, V]   = 16777216 floats
// Workspace layout (d_ws, needs ~1.05 MB):
//   logits [N] floats, then per-batch stats [B][8] floats
//     stats[b][0]=max, [1]=1/sum, [2..4]=pred xyz
// ---------------------------------------------------------------------------

#define EMBED_DIM 128
#define NB        32
#define NV        8192
#define NA        64
#define NTOT      (NB * NV)        // 262144
#define NTILES    (NTOT / 16)      // 16384

typedef __attribute__((ext_vector_type(16))) _Float16 v16h;
typedef __attribute__((ext_vector_type(8)))  float    v8f;

__device__ __forceinline__ v16h pack16_f16(const float4 a0, const float4 a1,
                                           const float4 a2, const float4 a3) {
  v16h r;
  r[0]  = (_Float16)a0.x; r[1]  = (_Float16)a0.y;
  r[2]  = (_Float16)a0.z; r[3]  = (_Float16)a0.w;
  r[4]  = (_Float16)a1.x; r[5]  = (_Float16)a1.y;
  r[6]  = (_Float16)a1.z; r[7]  = (_Float16)a1.w;
  r[8]  = (_Float16)a2.x; r[9]  = (_Float16)a2.y;
  r[10] = (_Float16)a2.z; r[11] = (_Float16)a2.w;
  r[12] = (_Float16)a3.x; r[13] = (_Float16)a3.y;
  r[14] = (_Float16)a3.z; r[15] = (_Float16)a3.w;
  return r;
}

// ---------------------------------------------------------------------------
// Kernel 1: logits[n] = dot(emb[n,:], W) + bias via v_wmma_f32_16x16x32_f16.
// One wave per 16-row tile; K=128 consumed as 4 chunks of 32.
// B fragment broadcasts W into all 16 columns, so every D column = logits.
// ---------------------------------------------------------------------------
__global__ __launch_bounds__(128) void logits_wmma_kernel(
    const float* __restrict__ emb, const float* __restrict__ Wv,
    const float* __restrict__ bias, float* __restrict__ logits) {
  const int lane = threadIdx.x & 31;
  const int wave = threadIdx.x >> 5;
  const int tile = blockIdx.x * 4 + wave;
  if (tile >= NTILES) return;  // uniform per wave: EXEC stays all-ones

  const int row = lane & 15;
  const int hi  = lane >> 4;     // 0 or 1 (half-wave)
  const int kA  = hi * 8;        // A-fragment K sub-offset
  const int kB  = hi * 8;        // B-fragment K sub-offset
  const float* rp = emb + (size_t)(tile * 16 + row) * EMBED_DIM;

  v8f acc = {};
#pragma unroll
  for (int c = 0; c < 4; ++c) {
    const int k0 = 32 * c;
    // A: 16x32 f16 fragment. Lane l holds row l&15,
    //    v[0..7] -> K = k0+kA+{0..7}, v[8..15] -> K = k0+16+kA+{0..7}.
    float4 a0 = *(const float4*)(rp + k0 + kA);
    float4 a1 = *(const float4*)(rp + k0 + kA + 4);
    float4 a2 = *(const float4*)(rp + k0 + 16 + kA);
    float4 a3 = *(const float4*)(rp + k0 + 16 + kA + 4);
    // B: 32x16 f16 fragment, every column = W. Lane l holds
    //    v[0..7] -> K = k0+kB+{0..7}, v[8..15] -> K = k0+16+kB+{0..7}.
    float4 b0 = *(const float4*)(Wv + k0 + kB);
    float4 b1 = *(const float4*)(Wv + k0 + kB + 4);
    float4 b2 = *(const float4*)(Wv + k0 + 16 + kB);
    float4 b3 = *(const float4*)(Wv + k0 + 16 + kB + 4);
    v16h A  = pack16_f16(a0, a1, a2, a3);
    v16h Bm = pack16_f16(b0, b1, b2, b3);
    acc = __builtin_amdgcn_wmma_f32_16x16x32_f16(
        /*neg_a=*/false, A, /*neg_b=*/false, Bm,
        /*c_mod=*/(short)0, acc, /*reuse_a=*/false, /*reuse_b=*/false);
  }

  const float bb = bias[0];
  // D layout: VGPR r = row r (lanes 0-15) / row r+8 (lanes 16-31); all
  // columns identical, so 16 lanes extract the 16 row logits.
  if ((lane & 8) == 0) {
    const int r = lane & 7;
    const int m = r + hi * 8;
    logits[tile * 16 + m] = acc[r] + bb;
  }
}

// ---------------------------------------------------------------------------
// Kernel 2: per-batch softmax stats + weighted coord sum. One block per batch.
// ---------------------------------------------------------------------------
__global__ __launch_bounds__(256) void softmax_stats_kernel(
    const float* __restrict__ logits, const float* __restrict__ coords,
    float* __restrict__ stats) {
  __shared__ float4 sred[256];
  const int b   = blockIdx.x;
  const int tid = threadIdx.x;
  const float* lg = logits + (size_t)b * NV;

  // pass 1: max
  float mx = -3.402823466e+38f;
  for (int v = tid; v < NV; v += 256) mx = fmaxf(mx, lg[v]);
  sred[tid].x = mx;
  __syncthreads();
  for (int off = 128; off > 0; off >>= 1) {
    if (tid < off) sred[tid].x = fmaxf(sred[tid].x, sred[tid + off].x);
    __syncthreads();
  }
  const float M = sred[0].x;
  __syncthreads();

  // pass 2: sum(exp) and sum(exp * coords)
  float s = 0.f, px = 0.f, py = 0.f, pz = 0.f;
  for (int v = tid; v < NV; v += 256) {
    const float e = __expf(lg[v] - M);
    const float* cp = coords + ((size_t)b * NV + v) * 3;
    s += e;
    px += e * cp[0];
    py += e * cp[1];
    pz += e * cp[2];
  }
  sred[tid] = make_float4(s, px, py, pz);
  __syncthreads();
  for (int off = 128; off > 0; off >>= 1) {
    if (tid < off) {
      const float4 x = sred[tid], y = sred[tid + off];
      sred[tid] = make_float4(x.x + y.x, x.y + y.y, x.z + y.z, x.w + y.w);
    }
    __syncthreads();
  }
  if (tid == 0) {
    const float4 t  = sred[0];
    const float inv = 1.0f / t.x;
    stats[b * 8 + 0] = M;
    stats[b * 8 + 1] = inv;
    stats[b * 8 + 2] = t.y * inv;
    stats[b * 8 + 3] = t.z * inv;
    stats[b * 8 + 4] = t.w * inv;
  }
}

// ---------------------------------------------------------------------------
// Kernel 3: attention_weights[b,a,v] = mask[b,a] * softmax(logits)[b,v].
// float4 per thread; 67 MB streaming write (the HBM-bound part).
// ---------------------------------------------------------------------------
__global__ __launch_bounds__(256) void attn_out_kernel(
    const float* __restrict__ logits, const float* __restrict__ stats,
    const int* __restrict__ mask, float* __restrict__ out_attn) {
  const int t  = blockIdx.x * 256 + threadIdx.x;
  const int v4 = t & (NV / 4 - 1);   // 0..2047
  const int ba = t >> 11;            // b*64 + a
  const int a  = ba & (NA - 1);
  const int b  = ba >> 6;

  const float m   = mask[b * NA + a] ? 1.0f : 0.0f;
  const float M   = stats[b * 8 + 0];
  const float inv = m * stats[b * 8 + 1];

  const float4 lg = *(const float4*)(logits + (size_t)b * NV + v4 * 4);
  float4 w;
  w.x = __expf(lg.x - M) * inv;
  w.y = __expf(lg.y - M) * inv;
  w.z = __expf(lg.z - M) * inv;
  w.w = __expf(lg.w - M) * inv;
  *(float4*)(out_attn + (size_t)ba * NV + v4 * 4) = w;
}

// ---------------------------------------------------------------------------
// Kernel 4: predicted_coords[b,a,c] = mask[b,a] * pred[b,c].
// ---------------------------------------------------------------------------
__global__ __launch_bounds__(256) void pred_out_kernel(
    const float* __restrict__ stats, const int* __restrict__ mask,
    float* __restrict__ out_pred) {
  const int idx = blockIdx.x * 256 + threadIdx.x;
  if (idx >= NB * NA * 3) return;
  const int c  = idx % 3;
  const int ba = idx / 3;
  const int a  = ba % NA;
  const int b  = ba / NA;
  out_pred[idx] = mask[b * NA + a] ? stats[b * 8 + 2 + c] : 0.0f;
}

extern "C" void kernel_launch(void* const* d_in, const int* in_sizes, int n_in,
                              void* d_out, int out_size, void* d_ws,
                              size_t ws_size, hipStream_t stream) {
  const float* emb    = (const float*)d_in[0];  // [N,128] f32
  const float* coords = (const float*)d_in[1];  // [N,3]   f32
  const float* Wv     = (const float*)d_in[2];  // [1,128] f32
  const float* bias   = (const float*)d_in[3];  // [1]     f32
  const int*   mask   = (const int*)d_in[4];    // [B,A]   bool->int
  // d_in[5] = batch_idx (contiguous equal groups by construction; unused)
  (void)in_sizes; (void)n_in; (void)out_size; (void)ws_size;

  float* logits = (float*)d_ws;           // N floats
  float* stats  = logits + NTOT;          // B*8 floats

  float* out_pred = (float*)d_out;               // B*A*3
  float* out_attn = out_pred + NB * NA * 3;      // B*A*V

  logits_wmma_kernel<<<NTILES / 4, 128, 0, stream>>>(emb, Wv, bias, logits);
  softmax_stats_kernel<<<NB, 256, 0, stream>>>(logits, coords, stats);
  attn_out_kernel<<<(NB * NA * (NV / 4)) / 256, 256, 0, stream>>>(
      logits, stats, mask, out_attn);
  pred_out_kernel<<<(NB * NA * 3 + 255) / 256, 256, 0, stream>>>(
      stats, mask, out_pred);
}
